// ExpertChoiceRouter_31258771980475
// MI455X (gfx1250) — matrix operand order repaired and checked by reading, not compiled, not run on hardware
//
#include <hip/hip_runtime.h>
#include <math.h>

// Problem dims (fixed by reference)
#define B_DIM 8
#define S_TOK 4096
#define D_DIM 4096
#define H_DIM 1024
#define M_TOT (B_DIM * S_TOK)   // 32768 tokens

// Tiling
#define M_TILE 128
#define N_TILE 64
#define K_TILE 32
#define KSTEPS (D_DIM / K_TILE)  // 128
#define LDSA_STRIDE 36           // 32 + pad, even (dword loads aligned)
#define LDSB_STRIDE 40           // 80-byte rows -> 16B-aligned B128 chunks
#define BTILE_ELEMS (N_TILE * K_TILE)  // 2048 bf16 per plane

typedef __attribute__((ext_vector_type(16))) __bf16       v16bf;
typedef __attribute__((ext_vector_type(8)))  float        v8f;
typedef __attribute__((ext_vector_type(8)))  unsigned int v8u;

__device__ __forceinline__ unsigned short bf16_bits(__bf16 h) {
  return __builtin_bit_cast(unsigned short, h);
}

// ---- gfx1250 async Global->LDS path (guarded; fallback = uint4 LDS store) ----
#if defined(__has_builtin)
#if __has_builtin(__builtin_amdgcn_global_load_async_to_lds_b128) && \
    __has_builtin(__builtin_amdgcn_s_wait_asynccnt)
#define USE_ASYNC_COPY 1
#endif
#endif
#ifndef USE_ASYNC_COPY
#define USE_ASYNC_COPY 0
#endif

#if USE_ASYNC_COPY
typedef int v4i_g __attribute__((vector_size(16)));              // gcc-style int4
typedef __attribute__((address_space(1))) v4i_g* g_v4i_ptr;      // global int4*
typedef __attribute__((address_space(3))) v4i_g* l_v4i_ptr;      // LDS int4*
#endif

__device__ __forceinline__ void copy16_g2lds(const void* g, void* l) {
#if USE_ASYNC_COPY
  __builtin_amdgcn_global_load_async_to_lds_b128(
      (g_v4i_ptr)(void*)g, (l_v4i_ptr)l, 0, 0);
#else
  *(uint4*)l = *(const uint4*)g;
#endif
}

__device__ __forceinline__ void drain_async_copies() {
#if USE_ASYNC_COPY
  __builtin_amdgcn_s_wait_asynccnt(0);
#endif
}

// ---------------------------------------------------------------------------
// Kernel 0a: zero the logits accumulator workspace
// ---------------------------------------------------------------------------
__global__ void zero_logits_kernel(float* __restrict__ logits) {
  int i = blockIdx.x * blockDim.x + threadIdx.x;
  if (i < M_TOT) logits[i] = 0.0f;
}

// ---------------------------------------------------------------------------
// Kernel 0b: one-shot W1 f32 -> bf16 hi/lo, written in per-(nblk,kt) tile
// layout: w1cvt[((nblk*KSTEPS + kt)*2 + plane)*2048 + n_local*32 + k_local]
// so the GEMM B staging is a pure contiguous 16B-per-lane copy.
// ---------------------------------------------------------------------------
__global__ __launch_bounds__(256)
void convert_w1_kernel(const float* __restrict__ W1,
                       unsigned short* __restrict__ w1cvt) {
  const int idx = blockIdx.x * blockDim.x + threadIdx.x;  // 0 .. D*H-1
  const int n = idx & (H_DIM - 1);
  const int k = idx >> 10;
  const float  x = W1[idx];
  const __bf16 h = (__bf16)x;
  const __bf16 l = (__bf16)(x - (float)h);
  const int nblk = n >> 6, nl = n & 63, kt = k >> 5, kl = k & 31;
  const size_t base = (size_t)((nblk * KSTEPS + kt) * 2) * BTILE_ELEMS;
  w1cvt[base + nl * K_TILE + kl]               = bf16_bits(h);
  w1cvt[base + BTILE_ELEMS + nl * K_TILE + kl] = bf16_bits(l);
}

// ---------------------------------------------------------------------------
// Kernel 1: fused  GEMM(hidden@W1) -> +b1 -> GELU(exact) -> *W2 -> atomic
//           partial-sum into logits[B*S].
// bf16 hi/lo split (3 WMMAs) for ~fp32 accuracy on v_wmma_f32_16x16x32_bf16.
// Block = 256 threads = 8 waves; wave w owns rows 16w..16w+15 x 64 cols.
// A tile double-buffered in registers; B tile async-copied from the
// pre-converted workspace directly into LDS.
// ---------------------------------------------------------------------------
__global__ __launch_bounds__(256)
void router_gemm_kernel(const float* __restrict__ hidden,
                        const unsigned short* __restrict__ w1cvt,
                        const float* __restrict__ b1,
                        const float* __restrict__ W2,
                        float* __restrict__ logits) {
  __shared__ unsigned short sAhi[M_TILE * LDSA_STRIDE];
  __shared__ unsigned short sAlo[M_TILE * LDSA_STRIDE];
  __shared__ unsigned short sBhi[N_TILE * LDSB_STRIDE];  // [n][k] transposed
  __shared__ unsigned short sBlo[N_TILE * LDSB_STRIDE];

  const int tid  = threadIdx.x;
  const int m0   = blockIdx.y * M_TILE;
  const int n0   = blockIdx.x * N_TILE;
  const int w    = tid >> 5;
  const int lane = tid & 31;
  const int half = lane >> 4;
  const int ln   = lane & 15;

  v8f acc[4];
#pragma unroll
  for (int i = 0; i < 4; ++i) {
#pragma unroll
    for (int j = 0; j < 8; ++j) acc[i][j] = 0.0f;
  }

  // --- preload A tile kt=0 into registers (software pipeline stage 0) ---
  float4 aCur[4];
#pragma unroll
  for (int i = 0; i < 4; ++i) {
    const int l = i * 256 + tid, row = l >> 3, c4 = l & 7;
    aCur[i] = *(const float4*)(hidden + (size_t)(m0 + row) * D_DIM + c4 * 4);
  }

  for (int kt = 0; kt < KSTEPS; ++kt) {
    const int k0 = kt * K_TILE;

    // ---- issue async B tile copy (hi+lo planes) from workspace -> LDS ----
    {
      const unsigned short* tileHi =
          w1cvt + (size_t)((blockIdx.x * KSTEPS + kt) * 2) * BTILE_ELEMS;
      const int nl = tid >> 2, part = tid & 3;     // 64 n x 4 chunks of 8 bf16
      const int go = nl * K_TILE + part * 8;
      const int lo = nl * LDSB_STRIDE + part * 8;
      copy16_g2lds(tileHi + go, &sBhi[lo]);
      copy16_g2lds(tileHi + BTILE_ELEMS + go, &sBlo[lo]);
    }

    // ---- convert registered A tile -> hi/lo bf16 in LDS ----
#pragma unroll
    for (int i = 0; i < 4; ++i) {
      const int l = i * 256 + tid, row = l >> 3, c4 = l & 7;
      const float xs[4] = {aCur[i].x, aCur[i].y, aCur[i].z, aCur[i].w};
#pragma unroll
      for (int j = 0; j < 4; ++j) {
        const float  x  = xs[j];
        const __bf16 h  = (__bf16)x;
        const __bf16 lo = (__bf16)(x - (float)h);
        sAhi[row * LDSA_STRIDE + c4 * 4 + j] = bf16_bits(h);
        sAlo[row * LDSA_STRIDE + c4 * 4 + j] = bf16_bits(lo);
      }
    }

    // ---- issue next A tile global loads (consumed next iteration) ----
    float4 aNxt[4];
    if (kt + 1 < KSTEPS) {
#pragma unroll
      for (int i = 0; i < 4; ++i) {
        const int l = i * 256 + tid, row = l >> 3, c4 = l & 7;
        aNxt[i] = *(const float4*)(hidden + (size_t)(m0 + row) * D_DIM +
                                   (k0 + K_TILE) + c4 * 4);
      }
      __builtin_prefetch(
          hidden + (size_t)(m0 + (tid & 127)) * D_DIM + k0 + 2 * K_TILE, 0, 1);
    }

    drain_async_copies();   // own wave's B copies landed in LDS
    __syncthreads();        // publish A stores + all waves' B copies

    // ---- A fragments (contiguous k order: k = 16*half + s, matches B) ----
    v8u ahi_u, alo_u;
    const int arow = 16 * w + ln;
#pragma unroll
    for (int v = 0; v < 8; ++v) {
      const int K0 = 16 * half + 2 * v;
      ahi_u[v] = *(const unsigned int*)&sAhi[arow * LDSA_STRIDE + K0];
      alo_u[v] = *(const unsigned int*)&sAlo[arow * LDSA_STRIDE + K0];
    }
    const v16bf a_hi = __builtin_bit_cast(v16bf, ahi_u);
    const v16bf a_lo = __builtin_bit_cast(v16bf, alo_u);

#pragma unroll
    for (int ns = 0; ns < 4; ++ns) {
      const int bcol = 16 * ns + ln;
      const int kb   = 16 * half;
      v8u bhi_u, blo_u;
#pragma unroll
      for (int v = 0; v < 8; ++v) {
        bhi_u[v] = *(const unsigned int*)&sBhi[bcol * LDSB_STRIDE + kb + 2 * v];
        blo_u[v] = *(const unsigned int*)&sBlo[bcol * LDSB_STRIDE + kb + 2 * v];
      }
      const v16bf b_hi = __builtin_bit_cast(v16bf, bhi_u);
      const v16bf b_lo = __builtin_bit_cast(v16bf, blo_u);

      acc[ns] = __builtin_amdgcn_wmma_f32_16x16x32_bf16(false, a_hi, false, b_hi,
                                                        (short)0, acc[ns], false, false);
      acc[ns] = __builtin_amdgcn_wmma_f32_16x16x32_bf16(false, a_hi, false, b_lo,
                                                        (short)0, acc[ns], false, false);
      acc[ns] = __builtin_amdgcn_wmma_f32_16x16x32_bf16(false, a_lo, false, b_hi,
                                                        (short)0, acc[ns], false, false);
    }
    __syncthreads();        // LDS reads done before next tile overwrites

    if (kt + 1 < KSTEPS) {
#pragma unroll
      for (int i = 0; i < 4; ++i) aCur[i] = aNxt[i];
    }
  }

  // ---- epilogue: +b1, exact GELU, *W2[n], reduce over n, atomic logits ----
  float rsum[8];
#pragma unroll
  for (int r = 0; r < 8; ++r) rsum[r] = 0.0f;

#pragma unroll
  for (int ns = 0; ns < 4; ++ns) {
    const int   gn = n0 + 16 * ns + ln;   // D-matrix: lane%16 = column
    const float w2 = W2[gn];
    const float bb = b1[gn];
#pragma unroll
    for (int r = 0; r < 8; ++r) {
      const float x = acc[ns][r] + bb;
      const float g = 0.5f * x * (1.0f + erff(x * 0.70710678118654752f));
      rsum[r] += g * w2;
    }
  }
#pragma unroll
  for (int r = 0; r < 8; ++r) {
#pragma unroll
    for (int mask = 1; mask < 16; mask <<= 1)
      rsum[r] += __shfl_xor(rsum[r], mask, 32);
  }
  if (ln == 0) {
#pragma unroll
    for (int r = 0; r < 8; ++r)
      atomicAdd(&logits[m0 + 16 * w + 8 * half + r], rsum[r]);
  }
}

// ---------------------------------------------------------------------------
// Kernel 2: per-row sigmoid, masking, variable-k top-k (exact JAX-stable
// ranks via counting), write router_weights then selected_mask (as float).
// ---------------------------------------------------------------------------
__global__ __launch_bounds__(256)
void router_select_kernel(const float* __restrict__ logits,
                          const unsigned char* __restrict__ active_mask,
                          const float* __restrict__ b2,
                          float* __restrict__ out) {
  const int b   = blockIdx.x;
  const int tid = threadIdx.x;
  __shared__ float m_arr[S_TOK];
  __shared__ int   s_active;

  if (tid == 0) s_active = 0;
  __syncthreads();

  const float bias = b2[0];
  int local_active = 0;
  for (int i = tid; i < S_TOK; i += 256) {
    const float logit = logits[b * S_TOK + i] + bias;
    const float score = 1.0f / (1.0f + expf(-logit));   // ALPHA == 1
    const bool  act   = active_mask[b * S_TOK + i] != 0;
    m_arr[i] = act ? score : -INFINITY;
    local_active += act ? 1 : 0;
  }
  atomicAdd(&s_active, local_active);
  __syncthreads();

  const int na = s_active;
  int k = (int)floorf(0.5f * (float)na);   // CAPACITY = 0.5
  k = max(k, 1);
  k = min(k, max(na, 1));
  if (na == 0) k = 0;

  for (int i = tid; i < S_TOK; i += 256) {
    const float mi = m_arr[i];
    int cnt = 0;
    for (int j = 0; j < S_TOK; ++j) {
      const float mj = m_arr[j];
      cnt += (mj > mi) || (mj == mi && j < i);   // stable argsort tie-break
    }
    const bool  sel = cnt < k;
    out[b * S_TOK + i]         = sel ? mi : 0.0f;          // router_weights
    out[M_TOT + b * S_TOK + i] = sel ? 1.0f : 0.0f;        // selected_mask
  }
}

// ---------------------------------------------------------------------------
extern "C" void kernel_launch(void* const* d_in, const int* in_sizes, int n_in,
                              void* d_out, int out_size, void* d_ws, size_t ws_size,
                              hipStream_t stream) {
  (void)in_sizes; (void)n_in; (void)out_size; (void)ws_size;
  const float*         hidden = (const float*)d_in[0];
  const unsigned char* amask  = (const unsigned char*)d_in[1];  // jnp.bool_ = 1 byte
  const float*         W1     = (const float*)d_in[2];
  const float*         b1     = (const float*)d_in[3];
  const float*         W2     = (const float*)d_in[4];
  const float*         b2     = (const float*)d_in[5];
  float*               out    = (float*)d_out;

  // workspace: [0, 128KB) logits f32; [128KB, 128KB+16MB) W1 bf16 hi/lo tiles
  float*          logits = (float*)d_ws;
  unsigned short* w1cvt  = (unsigned short*)((char*)d_ws + (size_t)M_TOT * 4);

  zero_logits_kernel<<<M_TOT / 256, 256, 0, stream>>>(logits);
  convert_w1_kernel<<<(D_DIM * H_DIM) / 256, 256, 0, stream>>>(W1, w1cvt);

  dim3 grid(H_DIM / N_TILE, M_TOT / M_TILE);   // x (N) fastest -> A tile L2 reuse
  router_gemm_kernel<<<grid, 256, 0, stream>>>(hidden, w1cvt, b1, W2, logits);

  router_select_kernel<<<B_DIM, 256, 0, stream>>>(logits, amask, b2, out);
}
